// SwinTransformerBlock_87582973100520
// MI455X (gfx1250) — matrix operand order repaired and checked by reading, not compiled, not run on hardware
//
#include <hip/hip_runtime.h>
#include <hip/hip_bf16.h>

// ---------------------------------------------------------------------------
// Swin Transformer block, fused, bf16 WMMA + async LDS copy (gfx1250, wave32)
// ---------------------------------------------------------------------------

#define WS_   8
#define SHIFT_ 4
#define NH_   8
#define HD_   32
#define DIM_  256
#define HID_  1024
#define B_    8
#define H_    128
#define W_    128
#define L_    (H_ * W_)          // 16384
#define NTOK_ (B_ * L_)          // 131072
#define NWIN_ (B_ * (H_/WS_) * (W_/WS_))  // 2048
#define EPS_  1e-5f

typedef __attribute__((ext_vector_type(16))) __bf16 v16bf;
typedef __attribute__((ext_vector_type(8)))  float  v8f;

#define WMMA_BF16(a, b, c) \
    __builtin_amdgcn_wmma_f32_16x16x32_bf16(false, (a), false, (b), (short)0, (c), false, false)

// ---- async global -> LDS copy (CDNA5 ASYNCcnt path) ------------------------
__device__ __forceinline__ void async_copy_b128(void* lds_dst, const void* gsrc) {
    const unsigned      lo = (unsigned)(unsigned long long)lds_dst;   // LDS byte offset
    const unsigned long long ga = (unsigned long long)gsrc;
    asm volatile("global_load_async_to_lds_b128 %0, %1, off"
                 :: "v"(lo), "v"(ga) : "memory");
}
__device__ __forceinline__ void async_wait0() {
    asm volatile("s_wait_asynccnt 0x0" ::: "memory");
}

// ---- fragment loaders (wave32; contiguous k = 16*(lane/16) + e in A and B) ----

// A: 16x32 bf16 tile, row-major, leading dim ld.
__device__ __forceinline__ v16bf frag_a(const __bf16* p, int ld, int lane) {
    v16bf a;
    const int m  = lane & 15;
    const int kb = (lane >> 4) << 4;          // 0 or 16
    const __bf16* row = p + m * ld;
#pragma unroll
    for (int e = 0; e < 16; ++e) a[e] = row[kb + e];
    return a;
}

// B: 32x16 where B[k][n] = w[n*ld + k]  (w is NxK row-major, i.e. W^T access)
__device__ __forceinline__ v16bf frag_b_nt(const __bf16* w, int ld, int lane) {
    v16bf b;
    const int n  = lane & 15;
    const int kb = (lane >> 4) << 4;          // 0 or 16
    const __bf16* col = w + n * ld + kb;
#pragma unroll
    for (int e = 0; e < 16; ++e) b[e] = col[e];
    return b;
}

// B: 32x16 where B[k][n] = m[k*ld + n]  (KxN row-major)
__device__ __forceinline__ v16bf frag_b_rm(const __bf16* mm, int ld, int lane) {
    v16bf b;
    const int n  = lane & 15;
    const int kb = (lane >> 4) << 4;
#pragma unroll
    for (int e = 0; e < 16; ++e) b[e] = mm[(kb + e) * ld + n];
    return b;
}

// half-wave (16-lane) reductions: a C-tile row lives in lanes sharing bit4
__device__ __forceinline__ float hmax16(float v) {
#pragma unroll
    for (int m = 1; m < 16; m <<= 1) v = fmaxf(v, __shfl_xor(v, m, 32));
    return v;
}
__device__ __forceinline__ float hsum16(float v) {
#pragma unroll
    for (int m = 1; m < 16; m <<= 1) v += __shfl_xor(v, m, 32);
    return v;
}
__device__ __forceinline__ float wsum32(float v) {
#pragma unroll
    for (int m = 1; m < 32; m <<= 1) v += __shfl_xor(v, m, 32);
    return v;
}

// ---------------------------------------------------------------------------
// Kernel 0: fp32 -> bf16 weight conversion (qkv_w | proj_w | fc1_w | fc2_w)
// ---------------------------------------------------------------------------
__global__ void k_wcvt(const float* qkvw, const float* projw,
                       const float* fc1w, const float* fc2w,
                       __bf16* qkvw_h, __bf16* projw_h,
                       __bf16* fc1w_h, __bf16* fc2w_h) {
    int i = blockIdx.x * blockDim.x + threadIdx.x;
    const int n_qkv = 3 * DIM_ * DIM_;
    const int n_prj = DIM_ * DIM_;
    const int n_fc1 = HID_ * DIM_;
    const int n_fc2 = DIM_ * HID_;
    if (i < n_qkv)                        { qkvw_h[i] = (__bf16)qkvw[i]; return; }
    i -= n_qkv;
    if (i < n_prj)                        { projw_h[i] = (__bf16)projw[i]; return; }
    i -= n_prj;
    if (i < n_fc1)                        { fc1w_h[i] = (__bf16)fc1w[i]; return; }
    i -= n_fc1;
    if (i < n_fc2)                        { fc2w_h[i] = (__bf16)fc2w[i]; return; }
}

// ---------------------------------------------------------------------------
// Kernel 1: LayerNorm1 -> bf16 xn  (one wave per token)
// ---------------------------------------------------------------------------
__global__ __launch_bounds__(256) void k_ln1(const float* __restrict__ x,
                                             const float* __restrict__ g,
                                             const float* __restrict__ bta,
                                             __bf16* __restrict__ xn) {
    const int wave = threadIdx.x >> 5;
    const int lane = threadIdx.x & 31;
    const int tok  = blockIdx.x * 8 + wave;
    const float* row = x + (size_t)tok * DIM_;

    float v[8];
    float s = 0.f;
#pragma unroll
    for (int e = 0; e < 8; ++e) { v[e] = row[e * 32 + lane]; s += v[e]; }
    const float mean = wsum32(s) * (1.f / DIM_);
    float q = 0.f;
#pragma unroll
    for (int e = 0; e < 8; ++e) { float d = v[e] - mean; q += d * d; }
    const float rstd = rsqrtf(wsum32(q) * (1.f / DIM_) + EPS_);

    __bf16* orow = xn + (size_t)tok * DIM_;
#pragma unroll
    for (int e = 0; e < 8; ++e) {
        int c = e * 32 + lane;
        orow[c] = (__bf16)((v[e] - mean) * rstd * g[c] + bta[c]);
    }
}

// ---------------------------------------------------------------------------
// Kernel 2: fused shifted-window attention (one workgroup per 8x8 window)
//   LDS: s_x 32KB | s_qkv 96KB | s_p 64KB  = 192KB dynamic
// ---------------------------------------------------------------------------
__global__ __launch_bounds__(256) void k_attn(const __bf16* __restrict__ xn,
                                              const __bf16* __restrict__ qkvw,
                                              const float* __restrict__ qkvb,
                                              const float* __restrict__ rpb,
                                              const __bf16* __restrict__ projw,
                                              const float* __restrict__ projb,
                                              const float* __restrict__ xin,
                                              float* __restrict__ xmid) {
    extern __shared__ char smem[];
    __bf16* s_x   = (__bf16*)smem;                       // 64 x 256  (reused as attn-out)
    __bf16* s_qkv = (__bf16*)(smem + 64 * 256 * 2);      // 64 x 768
    __bf16* s_p   = (__bf16*)(smem + 64 * 256 * 2 + 64 * 768 * 2); // 8 x 64 x 64

    const int wave = threadIdx.x >> 5;
    const int lane = threadIdx.x & 31;
    const int nl   = lane & 15;
    const int mb   = (lane >> 4) << 3;

    const int win = blockIdx.x;
    const int b   = win >> 8;
    const int wy  = (win & 255) >> 4;
    const int wx  = win & 15;

    __builtin_prefetch(qkvw, 0, 2);   // global_prefetch_b8 on the weight stream
    __builtin_prefetch(projw, 0, 2);

    // ---- async gather of shifted window tile into LDS (ASYNCcnt path) ----
#pragma unroll
    for (int it = 0; it < 8; ++it) {
        const int r    = it * 8 + (threadIdx.x >> 5);
        const int colb = (threadIdx.x & 31) * 8;             // 8 bf16 = 16B
        const int ty = r >> 3, tx = r & 7;
        const int gy = (wy * WS_ + ty + SHIFT_) & (H_ - 1);
        const int gx = (wx * WS_ + tx + SHIFT_) & (W_ - 1);
        const __bf16* src = xn + (((size_t)b * L_ + gy * W_ + gx) * DIM_ + colb);
        async_copy_b128(&s_x[r * DIM_ + colb], src);
    }
    async_wait0();
    __syncthreads();

    // ---- QKV = xw @ qkv_w^T + b  (64 x 768, K=256) -> s_qkv (bf16) --------
    for (int tile = wave; tile < 192; tile += 8) {
        const int mt = tile / 48, nt = tile % 48;
        v8f acc = {};
#pragma unroll
        for (int k = 0; k < 8; ++k) {
            v16bf a  = frag_a(s_x + (mt * 16) * DIM_ + k * 32, DIM_, lane);
            v16bf bf = frag_b_nt(qkvw + (size_t)(nt * 16) * DIM_ + k * 32, DIM_, lane);
            acc = WMMA_BF16(a, bf, acc);
        }
        const int n0 = nt * 16 + nl;
        const float bias = qkvb[n0];
#pragma unroll
        for (int r = 0; r < 8; ++r)
            s_qkv[(mt * 16 + mb + r) * 768 + n0] = (__bf16)(acc[r] + bias);
    }
    __syncthreads();

    // ---- per-head attention: wave == head ---------------------------------
    const int head = wave;
    const float scale = 0.1767766953f;   // 32^-0.5

    v8f logit[16];                        // 4 mt x 4 nt tiles of QK^T
#pragma unroll
    for (int mt = 0; mt < 4; ++mt)
#pragma unroll
        for (int nt = 0; nt < 4; ++nt) {
            v16bf a  = frag_a(s_qkv + (mt * 16) * 768 + head * HD_, 768, lane);
            v16bf bf = frag_b_nt(s_qkv + (nt * 16) * 768 + DIM_ + head * HD_, 768, lane);
            v8f z = {};
            logit[mt * 4 + nt] = WMMA_BF16(a, bf, z);
        }

    // scale + relative-position bias + row softmax (row lives in 16 lanes)
#pragma unroll
    for (int mt = 0; mt < 4; ++mt) {
#pragma unroll
        for (int r = 0; r < 8; ++r) {
            const int m = mt * 16 + mb + r;
            const int my = m >> 3, mx = m & 7;
            float vals[4];
            float rmax = -3.4e38f;
#pragma unroll
            for (int nt = 0; nt < 4; ++nt) {
                const int n = nt * 16 + nl;
                const int ny = n >> 3, nx = n & 7;
                const int rel = (my - ny + WS_ - 1) * (2 * WS_ - 1) + (mx - nx + WS_ - 1);
                float v = logit[mt * 4 + nt][r] * scale + rpb[rel * NH_ + head];
                vals[nt] = v;
                rmax = fmaxf(rmax, v);
            }
            rmax = hmax16(rmax);
            float ssum = 0.f;
#pragma unroll
            for (int nt = 0; nt < 4; ++nt) { vals[nt] = __expf(vals[nt] - rmax); ssum += vals[nt]; }
            ssum = hsum16(ssum);
            const float inv = 1.f / ssum;
#pragma unroll
            for (int nt = 0; nt < 4; ++nt)
                s_p[head * 4096 + m * 64 + nt * 16 + nl] = (__bf16)(vals[nt] * inv);
        }
    }

    // ---- out_h = P @ V  (64 x 32, K=64) -> reuse s_x as attn output -------
#pragma unroll
    for (int mt = 0; mt < 4; ++mt)
#pragma unroll
        for (int nt2 = 0; nt2 < 2; ++nt2) {
            v8f acc = {};
#pragma unroll
            for (int k = 0; k < 2; ++k) {
                v16bf a  = frag_a(s_p + head * 4096 + (mt * 16) * 64 + k * 32, 64, lane);
                v16bf bf = frag_b_rm(s_qkv + (k * 32) * 768 + 2 * DIM_ + head * HD_ + nt2 * 16, 768, lane);
                acc = WMMA_BF16(a, bf, acc);
            }
#pragma unroll
            for (int r = 0; r < 8; ++r)
                s_x[(mt * 16 + mb + r) * DIM_ + head * HD_ + nt2 * 16 + nl] = (__bf16)acc[r];
        }
    __syncthreads();

    // ---- proj (64 x 256, K=256) + residual, scatter with reverse shift ----
    for (int tile = wave; tile < 64; tile += 8) {
        const int mt = tile >> 4, nt = tile & 15;
        v8f acc = {};
#pragma unroll
        for (int k = 0; k < 8; ++k) {
            v16bf a  = frag_a(s_x + (mt * 16) * DIM_ + k * 32, DIM_, lane);
            v16bf bf = frag_b_nt(projw + (size_t)(nt * 16) * DIM_ + k * 32, DIM_, lane);
            acc = WMMA_BF16(a, bf, acc);
        }
        const int n0 = nt * 16 + nl;
        const float pb = projb[n0];
#pragma unroll
        for (int r = 0; r < 8; ++r) {
            const int m = mt * 16 + mb + r;
            const int ty = m >> 3, tx = m & 7;
            const int gy = (wy * WS_ + ty + SHIFT_) & (H_ - 1);
            const int gx = (wx * WS_ + tx + SHIFT_) & (W_ - 1);
            const size_t gi = ((size_t)b * L_ + gy * W_ + gx) * DIM_ + n0;
            xmid[gi] = xin[gi] + acc[r] + pb;
        }
    }
}

// ---------------------------------------------------------------------------
// Kernel 3: fused LN2 + MLP (GELU) + residual. 64 tokens per workgroup.
//   LDS: s_xn 32KB | s_h 128KB = 160KB dynamic
// ---------------------------------------------------------------------------
__global__ __launch_bounds__(256) void k_mlp(const float* __restrict__ xmid,
                                             const float* __restrict__ g,
                                             const float* __restrict__ bta,
                                             const __bf16* __restrict__ fc1w,
                                             const float* __restrict__ fc1b,
                                             const __bf16* __restrict__ fc2w,
                                             const float* __restrict__ fc2b,
                                             float* __restrict__ out) {
    extern __shared__ char smem[];
    __bf16* s_xn = (__bf16*)smem;                    // 64 x 256
    __bf16* s_h  = (__bf16*)(smem + 64 * 256 * 2);   // 64 x 1024

    const int wave = threadIdx.x >> 5;
    const int lane = threadIdx.x & 31;
    const int nl   = lane & 15;
    const int mb   = (lane >> 4) << 3;
    const size_t tb = (size_t)blockIdx.x * 64;

    __builtin_prefetch(fc1w, 0, 2);
    __builtin_prefetch(fc2w, 0, 2);

    // ---- LN2 into LDS bf16 (wave handles 8 tokens) ------------------------
#pragma unroll
    for (int i = 0; i < 8; ++i) {
        const int t = wave * 8 + i;
        const float* row = xmid + (tb + t) * DIM_;
        float v[8];
        float s = 0.f;
#pragma unroll
        for (int e = 0; e < 8; ++e) { v[e] = row[e * 32 + lane]; s += v[e]; }
        const float mean = wsum32(s) * (1.f / DIM_);
        float q = 0.f;
#pragma unroll
        for (int e = 0; e < 8; ++e) { float d = v[e] - mean; q += d * d; }
        const float rstd = rsqrtf(wsum32(q) * (1.f / DIM_) + EPS_);
#pragma unroll
        for (int e = 0; e < 8; ++e) {
            const int c = e * 32 + lane;
            s_xn[t * DIM_ + c] = (__bf16)((v[e] - mean) * rstd * g[c] + bta[c]);
        }
    }
    __syncthreads();

    // ---- h = gelu(xn @ fc1^T + b1)   (64 x 1024, K=256) -------------------
    for (int tile = wave; tile < 256; tile += 8) {
        const int mt = tile >> 6, nt = tile & 63;
        v8f acc = {};
#pragma unroll
        for (int k = 0; k < 8; ++k) {
            v16bf a  = frag_a(s_xn + (mt * 16) * DIM_ + k * 32, DIM_, lane);
            v16bf bf = frag_b_nt(fc1w + (size_t)(nt * 16) * DIM_ + k * 32, DIM_, lane);
            acc = WMMA_BF16(a, bf, acc);
        }
        const int n0 = nt * 16 + nl;
        const float b1 = fc1b[n0];
#pragma unroll
        for (int r = 0; r < 8; ++r) {
            const float u = acc[r] + b1;
            const float ge = 0.5f * u * (1.f + erff(u * 0.70710678f));   // exact GELU
            s_h[(mt * 16 + mb + r) * HID_ + n0] = (__bf16)ge;
        }
    }
    __syncthreads();

    // ---- y = h @ fc2^T + b2 + xmid   (64 x 256, K=1024) -------------------
    for (int tile = wave; tile < 64; tile += 8) {
        const int mt = tile >> 4, nt = tile & 15;
        v8f acc = {};
#pragma unroll
        for (int k = 0; k < 32; ++k) {
            v16bf a  = frag_a(s_h + (mt * 16) * HID_ + k * 32, HID_, lane);
            v16bf bf = frag_b_nt(fc2w + (size_t)(nt * 16) * HID_ + k * 32, HID_, lane);
            acc = WMMA_BF16(a, bf, acc);
        }
        const int n0 = nt * 16 + nl;
        const float b2 = fc2b[n0];
#pragma unroll
        for (int r = 0; r < 8; ++r) {
            const int m = mt * 16 + mb + r;
            const size_t gi = (tb + m) * DIM_ + n0;
            // final output: never re-read -> non-temporal store, keep L2 for xmid
            __builtin_nontemporal_store(xmid[gi] + acc[r] + b2, &out[gi]);
        }
    }
}

// ---------------------------------------------------------------------------
// Launch
// ---------------------------------------------------------------------------
extern "C" void kernel_launch(void* const* d_in, const int* in_sizes, int n_in,
                              void* d_out, int out_size, void* d_ws, size_t ws_size,
                              hipStream_t stream) {
    const float* x       = (const float*)d_in[0];
    const float* norm1_w = (const float*)d_in[1];
    const float* norm1_b = (const float*)d_in[2];
    const float* qkv_w   = (const float*)d_in[3];
    const float* qkv_b   = (const float*)d_in[4];
    const float* rpb     = (const float*)d_in[5];
    const float* proj_w  = (const float*)d_in[6];
    const float* proj_b  = (const float*)d_in[7];
    const float* norm2_w = (const float*)d_in[8];
    const float* norm2_b = (const float*)d_in[9];
    const float* fc1_w   = (const float*)d_in[10];
    const float* fc1_b   = (const float*)d_in[11];
    const float* fc2_w   = (const float*)d_in[12];
    const float* fc2_b   = (const float*)d_in[13];

    char* ws = (char*)d_ws;
    size_t off = 0;
    __bf16* xn_h   = (__bf16*)(ws + off); off += (size_t)NTOK_ * DIM_ * 2;       // 64 MB
    float*  xmid   = (float*) (ws + off); off += (size_t)NTOK_ * DIM_ * 4;       // 128 MB
    __bf16* qkvw_h = (__bf16*)(ws + off); off += (size_t)3 * DIM_ * DIM_ * 2;
    __bf16* projw_h= (__bf16*)(ws + off); off += (size_t)DIM_ * DIM_ * 2;
    __bf16* fc1w_h = (__bf16*)(ws + off); off += (size_t)HID_ * DIM_ * 2;
    __bf16* fc2w_h = (__bf16*)(ws + off); off += (size_t)DIM_ * HID_ * 2;

    // 0) weights -> bf16
    {
        const int total = 3*DIM_*DIM_ + DIM_*DIM_ + HID_*DIM_ + DIM_*HID_;
        k_wcvt<<<(total + 255) / 256, 256, 0, stream>>>(
            qkv_w, proj_w, fc1_w, fc2_w, qkvw_h, projw_h, fc1w_h, fc2w_h);
    }
    // 1) LN1 -> bf16 xn
    k_ln1<<<NTOK_ / 8, 256, 0, stream>>>(x, norm1_w, norm1_b, xn_h);

    // 2) fused shifted-window attention (+proj +residual)
    {
        const size_t lds = 64 * 256 * 2 + 64 * 768 * 2 + 8 * 64 * 64 * 2;   // 192 KB
        k_attn<<<NWIN_, 256, lds, stream>>>(xn_h, qkvw_h, qkv_b, rpb,
                                            projw_h, proj_b, x, xmid);
    }
    // 3) fused LN2 + MLP + residual -> d_out
    {
        const size_t lds = 64 * 256 * 2 + 64 * 1024 * 2;                    // 160 KB
        k_mlp<<<NTOK_ / 64, 256, lds, stream>>>(xmid, norm2_w, norm2_b,
                                                fc1w_h, fc1_b, fc2w_h, fc2_b,
                                                (float*)d_out);
    }
}
